// MultiDEQPDEBlock_84688165143268
// MI455X (gfx1250) — compile-verified, matrix operand order checked by reading, not run on hardware
//
#include <hip/hip_runtime.h>
#include <math.h>

// ---------------------------------------------------------------------------
// MI455X (gfx1250) fused DEQ-PDE solver.
// conv1 (4->32, 3x3) as implicit GEMM on v_wmma_f32_16x16x32_f16 (K=36 pad 64)
// conv2 (32->1, 3x3) on VALU (N=1 would waste 15/16 of a WMMA tile).
// Whole working set (~10MB) is L2-resident on MI455X (192MB L2).
// ---------------------------------------------------------------------------

typedef __attribute__((ext_vector_type(16))) _Float16 v16h;
typedef __attribute__((ext_vector_type(8)))  float    v8f;

#define HWD   256
#define NPIX  65536      // 256*256
#define HID   32
#define NB    8

__device__ __forceinline__ float gelu_f(float x) {
  return 0.5f * x * (1.0f + erff(x * 0.70710678118654752440f));
}
__device__ __forceinline__ float sigm_f(float x) {
  return 1.0f / (1.0f + __expf(-x));
}
__device__ __forceinline__ v8f wmma16(v16h a, v16h b, v8f c) {
  // v_wmma_f32_16x16x32_f16 : D = A(16x32 f16) * B(32x16 f16) + C(16x16 f32)
  return __builtin_amdgcn_wmma_f32_16x16x32_f16(false, a, false, b, (short)0, c,
                                                false, false);
}

// ---------------------------------------------------------------------------
// Pre-pack conv1 weights into the per-lane WMMA B-register image.
// Layout (CDNA5 ISA 7.12.2, 16-bit A/B): lane L holds col N=L%16,
// K in [kb,kb+8) U [16+kb,16+kb+8), kb = (L/16)*8, elem e: K = kb+e (e<8),
// 16+kb+e-8 (e>=8).  K index = inCh*9 + tap, K>=36 zero-padded to 64.
// w1p[ ((chunk*2+nhalf)*32 + lane)*16 + e ]
// ---------------------------------------------------------------------------
__global__ void prep_kernel(const float* __restrict__ w1,
                            _Float16* __restrict__ w1p,
                            int* __restrict__ flag, float* __restrict__ feat) {
  int tid = threadIdx.x;
  for (int id = tid; id < 2048; id += 256) {
    int e = id & 15, lane = (id >> 4) & 31, nhalf = (id >> 9) & 1, chunk = id >> 10;
    int N  = nhalf * 16 + (lane & 15);
    int kb = (lane >> 4) * 8;
    int K  = chunk * 32 + ((e < 8) ? (kb + e) : (16 + kb + (e - 8)));
    float v = 0.f;
    if (K < 36) { int c = K / 9, tap = K % 9; v = w1[(N * 4 + c) * 9 + tap]; }
    w1p[id] = (_Float16)v;
  }
  if (tid < 32) feat[tid] = 0.f;
  if (tid == 0) *flag = 0;
}

// z0 = boundary (= inp channel 0)
__global__ void initz_kernel(const float* __restrict__ inp, float* __restrict__ z0) {
  int idx = blockIdx.x * 256 + threadIdx.x;      // 2048 * 256 = NB*NPIX
  int b = idx >> 16, pix = idx & 65535;
  z0[idx] = inp[(b * 3 + 0) * NPIX + pix];
}

// ---------------------------------------------------------------------------
// Stabilizer: alpha = sigmoid(conv3x3(gelu(conv3x3([b,m,k]))))  (one-time)
// ---------------------------------------------------------------------------
__global__ __launch_bounds__(256)
void stab_kernel(const float* __restrict__ inp,
                 const float* __restrict__ w1, const float* __restrict__ b1,
                 const float* __restrict__ w2, const float* __restrict__ b2,
                 float* __restrict__ alpha) {
  int b = blockIdx.x >> 8, t = blockIdx.x & 255;
  int oy0 = (t >> 4) << 4, ox0 = (t & 15) << 4;
  int tid = threadIdx.x;

  __shared__ float sIn[3][20][20];
  __shared__ float sH[324][16];
  __shared__ float sW1[432];
  __shared__ float sB1[16];
  __shared__ float sW2[144];

  for (int i = tid; i < 432; i += 256) sW1[i] = w1[i];
  if (tid < 16) sB1[tid] = b1[tid];
  if (tid < 144) sW2[tid] = w2[tid];
  for (int i = tid; i < 1200; i += 256) {
    int c = i / 400, r = i % 400, yy = r / 20, xx = r % 20;
    int gy = oy0 - 2 + yy, gx = ox0 - 2 + xx;
    float v = 0.f;
    if (gy >= 0 && gy < HWD && gx >= 0 && gx < HWD)
      v = inp[(b * 3 + c) * NPIX + gy * HWD + gx];
    sIn[c][yy][xx] = v;
  }
  __syncthreads();

  for (int q = tid; q < 324; q += 256) {
    int hy = q / 18, hx = q % 18;
    #pragma unroll
    for (int oc = 0; oc < 16; ++oc) {
      float acc = sB1[oc];
      #pragma unroll
      for (int c = 0; c < 3; ++c)
        #pragma unroll
        for (int tap = 0; tap < 9; ++tap)
          acc += sIn[c][hy + tap / 3][hx + tap % 3] * sW1[(oc * 3 + c) * 9 + tap];
      sH[q][oc] = gelu_f(acc);
    }
  }
  __syncthreads();

  int ty = tid >> 4, tx = tid & 15;
  float acc = b2[0];
  #pragma unroll
  for (int tap = 0; tap < 9; ++tap) {
    const float* hr = &sH[(ty + tap / 3) * 18 + tx + tap % 3][0];
    #pragma unroll
    for (int c = 0; c < 16; ++c) acc += hr[c] * sW2[c * 9 + tap];
  }
  alpha[b * NPIX + (oy0 + ty) * HWD + ox0 + tx] = sigm_f(acc);
}

// ---------------------------------------------------------------------------
// spec_ctrl reductions: per-batch sum|b|, sum b^2, sum k, sum k^2
// ---------------------------------------------------------------------------
__global__ void featred_kernel(const float* __restrict__ inp, float* __restrict__ feat) {
  int b = blockIdx.x >> 5, chunk = blockIdx.x & 31;
  int tid = threadIdx.x;
  const float* bp = inp + (b * 3 + 0) * NPIX;
  const float* kp = inp + (b * 3 + 2) * NPIX;
  float s1 = 0.f, s2 = 0.f, s3 = 0.f, s4 = 0.f;
  int base = chunk * 2048 + tid * 8;
  #pragma unroll
  for (int j = 0; j < 8; ++j) {
    float bb = fabsf(bp[base + j]), kk = kp[base + j];
    s1 += bb; s2 += bb * bb; s3 += kk; s4 += kk * kk;
  }
  __shared__ float ls[4];
  if (tid < 4) ls[tid] = 0.f;
  __syncthreads();
  atomicAdd(&ls[0], s1); atomicAdd(&ls[1], s2);
  atomicAdd(&ls[2], s3); atomicAdd(&ls[3], s4);
  __syncthreads();
  if (tid < 4) atomicAdd(&feat[b * 4 + tid], ls[tid]);
}

__global__ void gamma_kernel(const float* __restrict__ feat,
                             const float* __restrict__ fw1, const float* __restrict__ fb1,
                             const float* __restrict__ fw2, const float* __restrict__ fb2,
                             float* __restrict__ gamma) {
  int b = threadIdx.x;
  if (b >= NB) return;
  const float n = 65536.f;
  float m1 = feat[b * 4 + 0] / n;
  float sd1 = sqrtf(fmaxf((feat[b * 4 + 1] - n * m1 * m1) / (n - 1.f), 0.f));
  float m2 = feat[b * 4 + 2] / n;
  float sd2 = sqrtf(fmaxf((feat[b * 4 + 3] - n * m2 * m2) / (n - 1.f), 0.f));
  float f[5] = {m1, sd1, m2, sd2, 1.f};
  float out = fb2[0];
  for (int j = 0; j < 32; ++j) {
    float h = fb1[j];
    #pragma unroll
    for (int i = 0; i < 5; ++i) h += fw1[j * 5 + i] * f[i];
    out += fw2[j] * gelu_f(h);
  }
  // cfl_scale = (32/256)^2 = 1/64
  gamma[b] = (0.5f + 1.5f * sigm_f(out)) * 0.015625f;
}

__global__ void zero_kernel(float* __restrict__ gn, float* __restrict__ diffsq) {
  int tid = threadIdx.x;
  if (tid < 64) gn[tid] = 0.f;
  if (tid < 8) diffsq[tid] = 0.f;
}

// ---------------------------------------------------------------------------
// Pass 1: conv1 via WMMA on a 16x16 output tile, accumulate GroupNorm stats.
// ---------------------------------------------------------------------------
__global__ __launch_bounds__(256)
void pass1_kernel(const float* __restrict__ inp, const float* __restrict__ zsrc,
                  const _Float16* __restrict__ w1p, const float* __restrict__ b1,
                  float* __restrict__ gn, const int* __restrict__ flag, int ignoreflag) {
  if (!ignoreflag && *flag) return;                 // converged: stats unused
  int b = blockIdx.x >> 8, t = blockIdx.x & 255;
  int oy0 = (t >> 4) << 4, ox0 = (t & 15) << 4;
  int tid = threadIdx.x;

  __shared__ _Float16 sIn[4][18][18];
  __shared__ float sStat[8];

  for (int i = tid; i < 1296; i += 256) {
    int c = i / 324, r = i % 324, yy = r / 18, xx = r % 18;
    int gy = oy0 - 1 + yy, gx = ox0 - 1 + xx;
    float v = 0.f;
    if (gy >= 0 && gy < HWD && gx >= 0 && gx < HWD)
      v = (c == 0) ? zsrc[b * NPIX + gy * HWD + gx]
                   : inp[(b * 3 + (c - 1)) * NPIX + gy * HWD + gx];
    sIn[c][yy][xx] = (_Float16)v;
  }
  if (tid < 8) sStat[tid] = 0.f;
  __syncthreads();

  int wv = tid >> 5, lane = tid & 31;
  int px = lane & 15, kb = (lane >> 4) * 8;
  const v16h* wp = (const v16h*)w1p;
  v16h b00 = wp[0 * 32 + lane], b01 = wp[1 * 32 + lane];
  v16h b10 = wp[2 * 32 + lane], b11 = wp[3 * 32 + lane];

  for (int t2 = wv; t2 < 16; t2 += 8) {            // each pixel-tile = one row
    int py = t2;
    v16h a0 = {}, a1 = {};
    #pragma unroll
    for (int e = 0; e < 16; ++e) {
      int K = (e < 8) ? (kb + e) : (16 + kb + (e - 8));
      { int c = K / 9, tap = K % 9;
        a0[e] = sIn[c][py + tap / 3][px + tap % 3]; }       // K<32 always valid
      int K1 = K + 32;
      _Float16 v1 = (_Float16)0.f;
      if (K1 < 36) { int c = K1 / 9, tap = K1 % 9;
        v1 = sIn[c][py + tap / 3][px + tap % 3]; }
      a1[e] = v1;
    }
    v8f acc0 = {}, acc1 = {};
    acc0 = wmma16(a0, b00, acc0); acc0 = wmma16(a1, b10, acc0);
    acc1 = wmma16(a0, b01, acc1); acc1 = wmma16(a1, b11, acc1);

    int ch0 = lane & 15, ch1 = 16 + (lane & 15);
    float s = 0.f, ss = 0.f;
    #pragma unroll
    for (int v = 0; v < 8; ++v) { float x = acc0[v] + b1[ch0]; s += x; ss += x * x; }
    atomicAdd(&sStat[(ch0 >> 3) * 2 + 0], s);
    atomicAdd(&sStat[(ch0 >> 3) * 2 + 1], ss);
    s = 0.f; ss = 0.f;
    #pragma unroll
    for (int v = 0; v < 8; ++v) { float x = acc1[v] + b1[ch1]; s += x; ss += x * x; }
    atomicAdd(&sStat[(ch1 >> 3) * 2 + 0], s);
    atomicAdd(&sStat[(ch1 >> 3) * 2 + 1], ss);
  }
  __syncthreads();
  if (tid < 8) atomicAdd(&gn[(b * 4 + (tid >> 1)) * 2 + (tid & 1)], sStat[tid]);
}

__global__ void statfin_kernel(const float* __restrict__ gn,
                               float* __restrict__ mu, float* __restrict__ rs) {
  int i = threadIdx.x;
  if (i >= 32) return;
  const float n = 8.f * 65536.f;
  float m = gn[i * 2] / n;
  float var = gn[i * 2 + 1] / n - m * m;
  mu[i] = m;
  rs[i] = rsqrtf(fmaxf(var, 0.f) + 1e-5f);
}

// ---------------------------------------------------------------------------
// Pass 2: recompute conv1 (with halo, WMMA) -> GN+GELU into LDS h tile ->
// conv2 (VALU) -> damped update + mask blend -> diff reduction.
// ---------------------------------------------------------------------------
__global__ __launch_bounds__(256)
void pass2_kernel(const float* __restrict__ inp, const float* __restrict__ zsrc,
                  float* __restrict__ zdst, const _Float16* __restrict__ w1p,
                  const float* __restrict__ b1,
                  const float* __restrict__ gnw, const float* __restrict__ gnb,
                  const float* __restrict__ mu, const float* __restrict__ rs,
                  const float* __restrict__ w2, const float* __restrict__ b2,
                  const float* __restrict__ alpha, const float* __restrict__ gamma,
                  float* __restrict__ diffsq, const int* __restrict__ flag, int mode) {
  int b = blockIdx.x >> 8, t = blockIdx.x & 255;
  int oy0 = (t >> 4) << 4, ox0 = (t & 15) << 4;
  int tid = threadIdx.x;
  int ty = tid >> 4, tx = tid & 15;
  int gy = oy0 + ty, gx = ox0 + tx;
  int zidx = b * NPIX + gy * HWD + gx;

  if (mode == 0 && *flag) { zdst[zidx] = zsrc[zidx]; return; }  // copy-through
  __builtin_prefetch(&zsrc[zidx], 0, 0);

  __shared__ _Float16 sIn[4][20][20];
  __shared__ float sH[336][HID];
  __shared__ float sW2[288];          // transposed: [tap][c]
  __shared__ float sRed;

  for (int i = tid; i < 288; i += 256) sW2[(i % 9) * 32 + (i / 9)] = w2[i];
  for (int i = tid; i < 1600; i += 256) {
    int c = i / 400, r = i % 400, yy = r / 20, xx = r % 20;
    int giy = oy0 - 2 + yy, gix = ox0 - 2 + xx;
    float v = 0.f;
    if (giy >= 0 && giy < HWD && gix >= 0 && gix < HWD)
      v = (c == 0) ? zsrc[b * NPIX + giy * HWD + gix]
                   : inp[(b * 3 + (c - 1)) * NPIX + giy * HWD + gix];
    sIn[c][yy][xx] = (_Float16)v;
  }
  if (tid == 0) sRed = 0.f;
  __syncthreads();

  int wv = tid >> 5, lane = tid & 31;
  int kb = (lane >> 4) * 8, mrow = (lane >> 4) * 8;
  const v16h* wp = (const v16h*)w1p;
  v16h b00 = wp[0 * 32 + lane], b01 = wp[1 * 32 + lane];
  v16h b10 = wp[2 * 32 + lane], b11 = wp[3 * 32 + lane];
  int ch0 = lane & 15, ch1 = 16 + (lane & 15);
  int g0 = b * 4 + (ch0 >> 3), g1 = b * 4 + (ch1 >> 3);
  float mu0 = mu[g0], rs0 = rs[g0], mu1 = mu[g1], rs1 = rs[g1];
  float sc0 = gnw[ch0], sh0 = gnb[ch0], sc1 = gnw[ch1], sh1 = gnb[ch1];
  float bb0 = b1[ch0], bb1 = b1[ch1];

  for (int tt = wv; tt < 21; tt += 8) {            // 324 h-points, 21 tiles of 16
    int q = tt * 16 + (lane & 15);
    int qq = (q > 323) ? 323 : q;
    int hy = qq / 18, hx = qq % 18;
    v16h a0 = {}, a1 = {};
    #pragma unroll
    for (int e = 0; e < 16; ++e) {
      int K = (e < 8) ? (kb + e) : (16 + kb + (e - 8));
      { int c = K / 9, tap = K % 9;
        a0[e] = sIn[c][hy + tap / 3][hx + tap % 3]; }
      int K1 = K + 32;
      _Float16 v1 = (_Float16)0.f;
      if (K1 < 36) { int c = K1 / 9, tap = K1 % 9;
        v1 = sIn[c][hy + tap / 3][hx + tap % 3]; }
      a1[e] = v1;
    }
    v8f acc0 = {}, acc1 = {};
    acc0 = wmma16(a0, b00, acc0); acc0 = wmma16(a1, b10, acc0);
    acc1 = wmma16(a0, b01, acc1); acc1 = wmma16(a1, b11, acc1);

    #pragma unroll
    for (int v = 0; v < 8; ++v) {
      int qs = tt * 16 + v + mrow;                 // D layout: M = v + 8*(lane/16)
      if (qs < 324) {
        sH[qs][ch0] = gelu_f((acc0[v] + bb0 - mu0) * rs0 * sc0 + sh0);
        sH[qs][ch1] = gelu_f((acc1[v] + bb1 - mu1) * rs1 * sc1 + sh1);
      }
    }
  }
  __syncthreads();

  float dz = b2[0];
  #pragma unroll
  for (int tap = 0; tap < 9; ++tap) {
    const float* hr = &sH[(ty + tap / 3) * 18 + tx + tap % 3][0];
    const float* wr = &sW2[tap * 32];
    #pragma unroll
    for (int c = 0; c < 32; ++c) dz += hr[c] * wr[c];
  }
  float zold = zsrc[zidx];
  float zn = zold + gamma[b] * alpha[zidx] * dz;
  float m = inp[(b * 3 + 1) * NPIX + gy * HWD + gx];
  float bd = inp[(b * 3 + 0) * NPIX + gy * HWD + gx];
  float znew = zn * (1.f - m) + bd * m;
  zdst[zidx] = znew;

  float d = znew - zold;
  atomicAdd(&sRed, (mode == 0) ? d * d : 0.f);
  __syncthreads();
  if (tid == 0 && mode == 0) atomicAdd(&diffsq[b], sRed);
}

__global__ void difffin_kernel(const float* __restrict__ diffsq, int* __restrict__ flag) {
  if (*flag) return;
  float s = 0.f;
  for (int b = 0; b < NB; ++b) s += sqrtf(diffsq[b]);
  if (s * 0.125f < 1e-4f) *flag = 1;
}

// ---------------------------------------------------------------------------
extern "C" void kernel_launch(void* const* d_in, const int* in_sizes, int n_in,
                              void* d_out, int out_size, void* d_ws, size_t ws_size,
                              hipStream_t stream) {
  (void)in_sizes; (void)n_in; (void)out_size; (void)ws_size;
  const float* inp     = (const float*)d_in[0];
  const float* core_w1 = (const float*)d_in[1];
  const float* core_b1 = (const float*)d_in[2];
  const float* gn_w    = (const float*)d_in[3];
  const float* gn_b    = (const float*)d_in[4];
  const float* core_w2 = (const float*)d_in[5];
  const float* core_b2 = (const float*)d_in[6];
  const float* stab_w1 = (const float*)d_in[7];
  const float* stab_b1 = (const float*)d_in[8];
  const float* stab_w2 = (const float*)d_in[9];
  const float* stab_b2 = (const float*)d_in[10];
  const float* fc1_w   = (const float*)d_in[11];
  const float* fc1_b   = (const float*)d_in[12];
  const float* fc2_w   = (const float*)d_in[13];
  const float* fc2_b   = (const float*)d_in[14];

  float* ws     = (float*)d_ws;
  float* z0     = ws;                    // 524288
  float* z1     = ws + 524288;           // 524288
  float* alpha  = ws + 1048576;          // 524288
  float* gamma  = ws + 1572864;          // 8
  float* feat   = ws + 1572872;          // 32
  float* gn     = ws + 1572904;          // 64
  float* mu     = ws + 1572968;          // 32
  float* rs     = ws + 1573000;          // 32
  float* diffsq = ws + 1573032;          // 8
  int*   flag   = (int*)(ws + 1573040);  // 1
  _Float16* w1p = (_Float16*)(ws + 1573056); // 2048 halves (32B-aligned)

  prep_kernel<<<1, 256, 0, stream>>>(core_w1, w1p, flag, feat);
  initz_kernel<<<2048, 256, 0, stream>>>(inp, z0);
  stab_kernel<<<2048, 256, 0, stream>>>(inp, stab_w1, stab_b1, stab_w2, stab_b2, alpha);
  featred_kernel<<<256, 256, 0, stream>>>(inp, feat);
  gamma_kernel<<<1, 32, 0, stream>>>(feat, fc1_w, fc1_b, fc2_w, fc2_b, gamma);

  for (int it = 0; it < 50; ++it) {
    float* zs = (it & 1) ? z1 : z0;
    float* zd = (it & 1) ? z0 : z1;
    zero_kernel<<<1, 128, 0, stream>>>(gn, diffsq);
    pass1_kernel<<<2048, 256, 0, stream>>>(inp, zs, w1p, core_b1, gn, flag, 0);
    statfin_kernel<<<1, 32, 0, stream>>>(gn, mu, rs);
    pass2_kernel<<<2048, 256, 0, stream>>>(inp, zs, zd, w1p, core_b1, gn_w, gn_b,
                                           mu, rs, core_w2, core_b2, alpha, gamma,
                                           diffsq, flag, 0);
    difffin_kernel<<<1, 1, 0, stream>>>(diffsq, flag);
  }
  // final core_f(z_star) -> d_out (z_star lives in z0 after 50 iterations)
  zero_kernel<<<1, 128, 0, stream>>>(gn, diffsq);
  pass1_kernel<<<2048, 256, 0, stream>>>(inp, z0, w1p, core_b1, gn, flag, 1);
  statfin_kernel<<<1, 32, 0, stream>>>(gn, mu, rs);
  pass2_kernel<<<2048, 256, 0, stream>>>(inp, z0, (float*)d_out, w1p, core_b1,
                                         gn_w, gn_b, mu, rs, core_w2, core_b2,
                                         alpha, gamma, diffsq, flag, 1);
}
